// SoftGroupSize_13151189861055
// MI455X (gfx1250) — compile-verified, matrix operand order checked by reading, not run on hardware
//
#include <hip/hip_runtime.h>
#include <math.h>

typedef __attribute__((ext_vector_type(8))) int v8i;

#define NPT 256           // min_npoint (harness fixes this at 256)
#define MED_RANK ((NPT - 1) / 2)   // 127

// One workgroup (256 threads = 8 wave32) per proposal.
__global__ __launch_bounds__(256) void softgroup_feats_kernel(
    const int*   __restrict__ prop_off,   // (NPROP+1)
    const int*   __restrict__ prop_idx,   // (NPROP*NPT, 2) flattened, col 1 = point id
    const float* __restrict__ pt_off,     // (N_POINTS, 3)
    float*       __restrict__ out)        // (NPROP, 4): mean, median, min, max
{
    __shared__ __align__(16) float rad[NPT];
    __shared__ float wsum[8], wmin[8], wmax[8];

    const int t    = threadIdx.x;
    const int p    = blockIdx.x;
    const int lane = t & 31;
    const int wave = t >> 5;

    // ---- gather + norm -------------------------------------------------
    const int start = prop_off[p];                    // = p*NPT, but read it
    const int pid   = prop_idx[2 * (start + t) + 1];
    const float* q  = pt_off + 3ll * (long long)pid;  // 12B row -> global_load_b96
    const float x = q[0], y = q[1], z = q[2];
    const float r = sqrtf(fmaf(x, x, fmaf(y, y, z * z)));
    rad[t] = r;

    // ---- mean / min / max: wave32 shuffle tree + cross-wave LDS --------
    float s = r, mn = r, mx = r;
    #pragma unroll
    for (int d = 16; d > 0; d >>= 1) {
        s  += __shfl_xor(s, d, 32);
        mn  = fminf(mn, __shfl_xor(mn, d, 32));
        mx  = fmaxf(mx, __shfl_xor(mx, d, 32));
    }
    if (lane == 0) { wsum[wave] = s; wmin[wave] = mn; wmax[wave] = mx; }
    __syncthreads();

    if (t == 0) {
        float S = 0.0f, MN = wmin[0], MX = wmax[0];
        #pragma unroll
        for (int w = 0; w < 8; ++w) {
            S += wsum[w];
            MN = fminf(MN, wmin[w]);
            MX = fmaxf(MX, wmax[w]);
        }
        out[4 * p + 0] = S * (1.0f / (float)NPT);
        out[4 * p + 2] = MN;
        out[4 * p + 3] = MX;
    }

    // ---- median: rank-select via V_WMMA_I32_16X16X64_IU8 ---------------
    // rank_i = sum_j [ (r_j, j) <lex (r_i, i) ]  ==  (cmp-matrix) x ones.
    v8i bones;
    #pragma unroll
    for (int v = 0; v < 8; ++v) bones[v] = 0x01010101;   // B = all-ones u8

    const int hi = (lane >> 4) & 1;   // which K-half this lane supplies
    const int m  = lane & 15;         // A row / D column within the 16-row tile

    #pragma unroll
    for (int grp = 0; grp < 2; ++grp) {               // 2 x 16 rows per wave
        const int rowBase = wave * 32 + grp * 16;
        const int i  = rowBase + m;
        const float ri = rad[i];

        v8i acc = {};                                  // C starts at 0
        #pragma unroll
        for (int c = 0; c < 4; ++c) {                  // 4 K-chunks of 64
            const int jb = c * 64;
            v8i a;
            #pragma unroll
            for (int v = 0; v < 8; ++v) {
                // 8-bit A layout: VGPR v holds K = ks..ks+3 for this lane half
                const int ks = (v >> 1) * 16 + hi * 8 + (v & 1) * 4;
                const float4 f = *(const float4*)&rad[jb + ks]; // ds_load_b128
                const float fv[4] = { f.x, f.y, f.z, f.w };
                unsigned pk = 0;
                #pragma unroll
                for (int b = 0; b < 4; ++b) {
                    const int j = jb + ks + b;
                    const unsigned bit =
                        (fv[b] < ri) || ((fv[b] == ri) && (j < i));
                    pk |= bit << (8 * b);
                }
                a[v] = (int)pk;
            }
            // (sgn_a, A, sgn_b, B, C, reuse_a, reuse_b) -> i32 16x16 ranks
            acc = __builtin_amdgcn_wmma_i32_16x16x64_iu8(
                      false, a, false, bones, acc, false, false);
        }

        // D layout: lanes 0-15 col=lane, rows v+0..7; lanes 16-31 rows v+8..15.
        // Read column 0 (lanes 0 and 16): exactly one row has rank == 127.
        if (m == 0) {
            #pragma unroll
            for (int v = 0; v < 8; ++v) {
                const int row = rowBase + hi * 8 + v;
                if (acc[v] == MED_RANK) {
                    out[4 * p + 1] = rad[row];
                }
            }
        }
    }
}

extern "C" void kernel_launch(void* const* d_in, const int* in_sizes, int n_in,
                              void* d_out, int out_size, void* d_ws, size_t ws_size,
                              hipStream_t stream) {
    // inputs: 0=semantic_scores(f32, unused), 1=proposals_offset(i32),
    //         2=proposals_idx(i32, Nx2), 3=pt_offsets(f32, Nx3), 4=min_npoint(i32)
    const int*   prop_off = (const int*)d_in[1];
    const int*   prop_idx = (const int*)d_in[2];
    const float* pt_off   = (const float*)d_in[3];
    float*       out      = (float*)d_out;

    const int nprop = out_size / 4;   // 4096 proposals, 4 feats each
    (void)in_sizes; (void)n_in; (void)d_ws; (void)ws_size;

    softgroup_feats_kernel<<<nprop, NPT, 0, stream>>>(prop_off, prop_idx, pt_off, out);
}
